// EncodeProcessDecode_20083267076599
// MI455X (gfx1250) — compile-verified, matrix operand order checked by reading, not run on hardware
//
#include <hip/hip_runtime.h>

#define NN 25000
#define NE 400000
#define NODE_F 16
#define EDGE_F 4
#define WAVES 4
#define TPW 8   // tiles per wave per launch (amortize weight staging)
#define LDH 72  // padded leading dim for 64-wide LDS tiles (dword stride 36, gcd(36,64)=4)

typedef __bf16 v16bf __attribute__((ext_vector_type(16)));
typedef float  v8f   __attribute__((ext_vector_type(8)));

union FragBF {
    v16bf v;
    uint4 q[2];
    unsigned short u16[16];
};

// round-to-nearest-even fp32 -> bf16 (as raw u16)
__device__ inline unsigned short f2bf(float f) {
    unsigned u = __builtin_bit_cast(unsigned, f);
    unsigned r = u + 0x7FFFu + ((u >> 16) & 1u);
    return (unsigned short)(r >> 16);
}

// Load a 16-half WMMA operand fragment from an LDS tile [rows][ld] (bf16,
// row-major). "row" is M for A or N for the transposed-in-LDS B. kBase must
// already include the +8*hi lane-half offset. Both 16B halves are aligned by
// construction (ld % 8 == 0, kBase % 8 == 0) -> exactly 2x ds_load_b128.
// Element map: u16[0..7] = T[row][kBase..kBase+7], u16[8..15] = T[row][kBase+16..kBase+23],
// matching the CDNA5 16-bit A/B VGPR layout (K pairs 0-7 then 16-23; +8 via hi-half kBase).
__device__ inline v16bf load_frag(const unsigned short* t, int row, int ld, int kBase) {
    const unsigned short* p = t + row * ld + kBase;
    FragBF f;
    f.q[0] = *(const uint4*)(p);
    f.q[1] = *(const uint4*)(p + 16);
    return f.v;
}

// Cooperative fp32->bf16 weight load, transposed into LDS: dst[n*ldp + k] = W[k][n]
// n-fastest iteration order => coalesced global reads.
__device__ inline void load_wt(unsigned short* dst, const float* __restrict__ W,
                               int ldp, int realK, int ldn, int realN, int rows) {
    for (int i = threadIdx.x; i < rows * ldp; i += blockDim.x) {
        int k = i / rows, n = i % rows;
        float v = (k < realK && n < realN) ? W[(size_t)k * ldn + n] : 0.0f;
        dst[n * ldp + k] = f2bf(v);
    }
}

__device__ inline void load_bias(float* dst, const float* __restrict__ b, int realN) {
    for (int i = threadIdx.x; i < 64; i += blockDim.x)
        dst[i] = (i < realN) ? b[i] : 0.0f;
}

// Fused 3-layer MLP on one 16-row tile (per wave): KIN -> 64 -> 64 -> NTO*16.
// sIn: [16][LDIN] bf16 staged input (per-wave). sH: [16][LDH] per-wave scratch.
// Weights transposed in LDS: sW0t [64][LDIN], sW1t [64][LDH], sW2t [NTO*16][LDH].
template <int KIN, int LDIN, int NTO, bool LN>
__device__ inline void mlp3(const unsigned short* sIn, unsigned short* sH,
                            const unsigned short* sW0t, const unsigned short* sW1t,
                            const unsigned short* sW2t,
                            const float* sB0, const float* sB1, const float* sB2,
                            v8f (&o)[NTO]) {
    const int lane = threadIdx.x & 31;
    const int m    = lane & 15;
    const int hi   = lane >> 4;
    const int koff = 8 * hi;
    const v8f vzero = {0.f, 0.f, 0.f, 0.f, 0.f, 0.f, 0.f, 0.f};

    // ---- layer 0: [16 x KIN] @ [KIN x 64] ----
    v8f h[4];
#pragma unroll
    for (int nt = 0; nt < 4; ++nt) h[nt] = vzero;
#pragma unroll
    for (int kc = 0; kc < KIN; kc += 32) {
        v16bf a = load_frag(sIn, m, LDIN, kc + koff);
#pragma unroll
        for (int nt = 0; nt < 4; ++nt) {
            v16bf b = load_frag(sW0t, nt * 16 + m, LDIN, kc + koff);
            h[nt] = __builtin_amdgcn_wmma_f32_16x16x32_bf16(
                        false, a, false, b, (short)0, h[nt], false, false);
        }
    }
    // bias + relu, restage as bf16 (C-layout scatter: row = v + 8*hi, col = nt*16+m)
#pragma unroll
    for (int nt = 0; nt < 4; ++nt) {
        int c = nt * 16 + m;
        float bb = sB0[c];
#pragma unroll
        for (int v = 0; v < 8; ++v)
            sH[(v + koff) * LDH + c] = f2bf(fmaxf(h[nt][v] + bb, 0.0f));
    }
    __syncthreads();

    // ---- layer 1: [16 x 64] @ [64 x 64] ----
    v8f h2[4];
#pragma unroll
    for (int nt = 0; nt < 4; ++nt) h2[nt] = vzero;
#pragma unroll
    for (int kc = 0; kc < 64; kc += 32) {
        v16bf a = load_frag(sH, m, LDH, kc + koff);
#pragma unroll
        for (int nt = 0; nt < 4; ++nt) {
            v16bf b = load_frag(sW1t, nt * 16 + m, LDH, kc + koff);
            h2[nt] = __builtin_amdgcn_wmma_f32_16x16x32_bf16(
                         false, a, false, b, (short)0, h2[nt], false, false);
        }
    }
    __syncthreads();   // all reads of sH done before overwrite
#pragma unroll
    for (int nt = 0; nt < 4; ++nt) {
        int c = nt * 16 + m;
        float bb = sB1[c];
#pragma unroll
        for (int v = 0; v < 8; ++v)
            sH[(v + koff) * LDH + c] = f2bf(fmaxf(h2[nt][v] + bb, 0.0f));
    }
    __syncthreads();

    // ---- layer 2: [16 x 64] @ [64 x NTO*16] ----
#pragma unroll
    for (int nt = 0; nt < NTO; ++nt) o[nt] = vzero;
#pragma unroll
    for (int kc = 0; kc < 64; kc += 32) {
        v16bf a = load_frag(sH, m, LDH, kc + koff);
#pragma unroll
        for (int nt = 0; nt < NTO; ++nt) {
            v16bf b = load_frag(sW2t, nt * 16 + m, LDH, kc + koff);
            o[nt] = __builtin_amdgcn_wmma_f32_16x16x32_bf16(
                        false, a, false, b, (short)0, o[nt], false, false);
        }
    }
#pragma unroll
    for (int nt = 0; nt < NTO; ++nt) {
        float bb = sB2[nt * 16 + m];
#pragma unroll
        for (int v = 0; v < 8; ++v) o[nt][v] += bb;
    }

    if constexpr (LN) {
        // LayerNorm over 64 cols: row m=v+8*hi spans the 16-lane half-wave
#pragma unroll
        for (int v = 0; v < 8; ++v) {
            float s  = o[0][v] + o[1][v] + o[2][v] + o[3][v];
            float sq = o[0][v]*o[0][v] + o[1][v]*o[1][v] + o[2][v]*o[2][v] + o[3][v]*o[3][v];
#pragma unroll
            for (int msk = 1; msk <= 8; msk <<= 1) {
                s  += __shfl_xor(s,  msk, 32);
                sq += __shfl_xor(sq, msk, 32);
            }
            float mu  = s * 0.015625f;
            float var = sq * 0.015625f - mu * mu;
            float r   = rsqrtf(var + 1e-5f);
#pragma unroll
            for (int nt = 0; nt < 4; ++nt) o[nt][v] = (o[nt][v] - mu) * r;
        }
    }
}

// ---------------- encoder: rows x INF -> 64, LN; writes enc copy + state copy
template <int INF>
__global__ __launch_bounds__(128) void encoder_kernel(
    const float* __restrict__ in,
    const float* __restrict__ w0, const float* __restrict__ b0,
    const float* __restrict__ w1, const float* __restrict__ b1,
    const float* __restrict__ w2, const float* __restrict__ b2,
    float* __restrict__ out_enc, float* __restrict__ out_state, int nRows) {
    constexpr int KP = 32, LDIN = KP + 8;
    __shared__ __align__(16) unsigned short sW0t[64 * LDIN];
    __shared__ __align__(16) unsigned short sW1t[64 * LDH];
    __shared__ __align__(16) unsigned short sW2t[64 * LDH];
    __shared__ float sB0[64], sB1[64], sB2[64];
    __shared__ __align__(16) unsigned short sIn[WAVES][16 * LDIN];
    __shared__ __align__(16) unsigned short sH[WAVES][16 * LDH];

    load_wt(sW0t, w0, LDIN, INF, 64, 64, 64);
    load_wt(sW1t, w1, LDH, 64, 64, 64, 64);
    load_wt(sW2t, w2, LDH, 64, 64, 64, 64);
    load_bias(sB0, b0, 64); load_bias(sB1, b1, 64); load_bias(sB2, b2, 64);

    const int wv = threadIdx.x >> 5, lane = threadIdx.x & 31;
    const int hi = lane >> 4, koff = 8 * hi, nc = lane & 15;
    unsigned short* myIn = sIn[wv];

#pragma unroll 1
    for (int t = 0; t < TPW; ++t) {
        const int rowBase = ((blockIdx.x * TPW + t) * WAVES + wv) * 16;
        for (int i = lane; i < 16 * KP; i += 32) {
            int r = i / KP, c = i % KP;
            int gr = rowBase + r;
            float v = (gr < nRows && c < INF) ? in[(size_t)gr * INF + c] : 0.0f;
            myIn[r * LDIN + c] = f2bf(v);
        }
        __syncthreads();

        v8f o[4];
        mlp3<KP, LDIN, 4, true>(myIn, sH[wv], sW0t, sW1t, sW2t, sB0, sB1, sB2, o);

#pragma unroll
        for (int v = 0; v < 8; ++v) {
            int gr = rowBase + v + koff;
            if (gr < nRows) {
#pragma unroll
                for (int nt = 0; nt < 4; ++nt) {
                    size_t idx = (size_t)gr * 64 + nt * 16 + nc;
                    float val = o[nt][v];
                    out_enc[idx] = val;
                    out_state[idx] = val;
                }
            }
        }
    }
}

// ---------------- edge block: concat(pe, px[recv], px[send]) -> MLP(192) + LN
// pe updated in place (pe = e_enc + ue); ue scatter-added into agg.
__global__ __launch_bounds__(128) void edge_block_kernel(
    float* __restrict__ pe, const float* __restrict__ px,
    const float* __restrict__ e_enc,
    const int* __restrict__ recv, const int* __restrict__ send,
    float* __restrict__ agg,
    const float* __restrict__ w0, const float* __restrict__ b0,
    const float* __restrict__ w1, const float* __restrict__ b1,
    const float* __restrict__ w2, const float* __restrict__ b2) {
    constexpr int KP = 192, LDIN = KP + 8;
    __shared__ __align__(16) unsigned short sW0t[64 * LDIN];
    __shared__ __align__(16) unsigned short sW1t[64 * LDH];
    __shared__ __align__(16) unsigned short sW2t[64 * LDH];
    __shared__ float sB0[64], sB1[64], sB2[64];
    __shared__ __align__(16) unsigned short sIn[WAVES][16 * LDIN];
    __shared__ __align__(16) unsigned short sH[WAVES][16 * LDH];

    load_wt(sW0t, w0, LDIN, 192, 64, 64, 64);
    load_wt(sW1t, w1, LDH, 64, 64, 64, 64);
    load_wt(sW2t, w2, LDH, 64, 64, 64, 64);
    load_bias(sB0, b0, 64); load_bias(sB1, b1, 64); load_bias(sB2, b2, 64);

    const int wv = threadIdx.x >> 5, lane = threadIdx.x & 31;
    const int hi = lane >> 4, koff = 8 * hi, nc = lane & 15;
    unsigned short* myIn = sIn[wv];

#pragma unroll 1
    for (int t = 0; t < TPW; ++t) {
        const int eBase = ((blockIdx.x * TPW + t) * WAVES + wv) * 16;
        // gather stage: 16 edges x 48 float4 chunks, pack to bf16
        for (int u = lane; u < 16 * 48; u += 32) {
            int r = u / 48, q = u % 48;
            int e = eBase + r;
            int col = q * 4;
            float4 val = make_float4(0.f, 0.f, 0.f, 0.f);
            if (e < NE) {
                const float* src;
                if (col < 64) {
                    src = pe + (size_t)e * 64 + col;
                    if (e + WAVES * 16 < NE)  // prefetch next tile's pe rows
                        __builtin_prefetch(pe + (size_t)(e + WAVES * 16) * 64 + col, 0, 0);
                } else if (col < 128) {
                    src = px + (size_t)recv[e] * 64 + (col - 64);
                } else {
                    src = px + (size_t)send[e] * 64 + (col - 128);
                }
                val = *(const float4*)src;
            }
            unsigned int lo = (unsigned)f2bf(val.x) | ((unsigned)f2bf(val.y) << 16);
            unsigned int hh = (unsigned)f2bf(val.z) | ((unsigned)f2bf(val.w) << 16);
            unsigned int* d = (unsigned int*)(myIn + r * LDIN + col);
            d[0] = lo; d[1] = hh;
        }
        __syncthreads();

        v8f o[4];
        mlp3<KP, LDIN, 4, true>(myIn, sH[wv], sW0t, sW1t, sW2t, sB0, sB1, sB2, o);

#pragma unroll
        for (int v = 0; v < 8; ++v) {
            int e = eBase + v + koff;
            if (e < NE) {
                int re = recv[e];
#pragma unroll
                for (int nt = 0; nt < 4; ++nt) {
                    float ue = o[nt][v];
                    size_t idx = (size_t)e * 64 + nt * 16 + nc;
                    pe[idx] = e_enc[idx] + ue;
                    atomicAdd(&agg[(size_t)re * 64 + nt * 16 + nc], ue);
                }
            }
        }
    }
}

// ---------------- node block: concat(px, agg) -> MLP(128) + LN; px = x_enc + un
__global__ __launch_bounds__(128) void node_block_kernel(
    float* __restrict__ px, const float* __restrict__ agg,
    const float* __restrict__ x_enc,
    const float* __restrict__ w0, const float* __restrict__ b0,
    const float* __restrict__ w1, const float* __restrict__ b1,
    const float* __restrict__ w2, const float* __restrict__ b2, int nRows) {
    constexpr int KP = 128, LDIN = KP + 8;
    __shared__ __align__(16) unsigned short sW0t[64 * LDIN];
    __shared__ __align__(16) unsigned short sW1t[64 * LDH];
    __shared__ __align__(16) unsigned short sW2t[64 * LDH];
    __shared__ float sB0[64], sB1[64], sB2[64];
    __shared__ __align__(16) unsigned short sIn[WAVES][16 * LDIN];
    __shared__ __align__(16) unsigned short sH[WAVES][16 * LDH];

    load_wt(sW0t, w0, LDIN, 128, 64, 64, 64);
    load_wt(sW1t, w1, LDH, 64, 64, 64, 64);
    load_wt(sW2t, w2, LDH, 64, 64, 64, 64);
    load_bias(sB0, b0, 64); load_bias(sB1, b1, 64); load_bias(sB2, b2, 64);

    const int wv = threadIdx.x >> 5, lane = threadIdx.x & 31;
    const int hi = lane >> 4, koff = 8 * hi, nc = lane & 15;
    unsigned short* myIn = sIn[wv];

#pragma unroll 1
    for (int t = 0; t < TPW; ++t) {
        const int rowBase = ((blockIdx.x * TPW + t) * WAVES + wv) * 16;
        for (int u = lane; u < 16 * 32; u += 32) {
            int r = u / 32, q = u % 32;
            int gr = rowBase + r;
            int col = q * 4;
            float4 val = make_float4(0.f, 0.f, 0.f, 0.f);
            if (gr < nRows) {
                const float* src = (col < 64) ? (px + (size_t)gr * 64 + col)
                                              : (agg + (size_t)gr * 64 + (col - 64));
                val = *(const float4*)src;
            }
            unsigned int lo = (unsigned)f2bf(val.x) | ((unsigned)f2bf(val.y) << 16);
            unsigned int hh = (unsigned)f2bf(val.z) | ((unsigned)f2bf(val.w) << 16);
            unsigned int* d = (unsigned int*)(myIn + r * LDIN + col);
            d[0] = lo; d[1] = hh;
        }
        __syncthreads();

        v8f o[4];
        mlp3<KP, LDIN, 4, true>(myIn, sH[wv], sW0t, sW1t, sW2t, sB0, sB1, sB2, o);

#pragma unroll
        for (int v = 0; v < 8; ++v) {
            int gr = rowBase + v + koff;
            if (gr < nRows) {
#pragma unroll
                for (int nt = 0; nt < 4; ++nt) {
                    size_t idx = (size_t)gr * 64 + nt * 16 + nc;
                    px[idx] = x_enc[idx] + o[nt][v];
                }
            }
        }
    }
}

// ---------------- decoder: 64 -> 64 -> 64 -> 3 (no LN)
__global__ __launch_bounds__(128) void decoder_kernel(
    const float* __restrict__ px, float* __restrict__ out,
    const float* __restrict__ w0, const float* __restrict__ b0,
    const float* __restrict__ w1, const float* __restrict__ b1,
    const float* __restrict__ w2, const float* __restrict__ b2, int nRows) {
    constexpr int KP = 64, LDIN = KP + 8;
    __shared__ __align__(16) unsigned short sW0t[64 * LDIN];
    __shared__ __align__(16) unsigned short sW1t[64 * LDH];
    __shared__ __align__(16) unsigned short sW2t[16 * LDH];
    __shared__ float sB0[64], sB1[64], sB2[64];
    __shared__ __align__(16) unsigned short sIn[WAVES][16 * LDIN];
    __shared__ __align__(16) unsigned short sH[WAVES][16 * LDH];

    load_wt(sW0t, w0, LDIN, 64, 64, 64, 64);
    load_wt(sW1t, w1, LDH, 64, 64, 64, 64);
    load_wt(sW2t, w2, LDH, 64, 3, 3, 16);   // 64x3 weights, padded to 16 cols
    load_bias(sB0, b0, 64); load_bias(sB1, b1, 64); load_bias(sB2, b2, 3);

    const int wv = threadIdx.x >> 5, lane = threadIdx.x & 31;
    const int hi = lane >> 4, koff = 8 * hi, nc = lane & 15;
    unsigned short* myIn = sIn[wv];

#pragma unroll 1
    for (int t = 0; t < TPW; ++t) {
        const int rowBase = ((blockIdx.x * TPW + t) * WAVES + wv) * 16;
        for (int u = lane; u < 16 * 16; u += 32) {
            int r = u / 16, q = u % 16;
            int gr = rowBase + r;
            int col = q * 4;
            float4 val = make_float4(0.f, 0.f, 0.f, 0.f);
            if (gr < nRows) val = *(const float4*)(px + (size_t)gr * 64 + col);
            unsigned int lo = (unsigned)f2bf(val.x) | ((unsigned)f2bf(val.y) << 16);
            unsigned int hh = (unsigned)f2bf(val.z) | ((unsigned)f2bf(val.w) << 16);
            unsigned int* d = (unsigned int*)(myIn + r * LDIN + col);
            d[0] = lo; d[1] = hh;
        }
        __syncthreads();

        v8f o[1];
        mlp3<KP, LDIN, 1, false>(myIn, sH[wv], sW0t, sW1t, sW2t, sB0, sB1, sB2, o);

        if (nc < 3) {
#pragma unroll
            for (int v = 0; v < 8; ++v) {
                int gr = rowBase + v + koff;
                if (gr < nRows) out[(size_t)gr * 3 + nc] = o[0][v];
            }
        }
    }
}

__global__ void zero_kernel(float* __restrict__ p, int n) {
    int i = blockIdx.x * blockDim.x + threadIdx.x;
    if (i < n) p[i] = 0.0f;
}

extern "C" void kernel_launch(void* const* d_in, const int* in_sizes, int n_in,
                              void* d_out, int out_size, void* d_ws, size_t ws_size,
                              hipStream_t stream) {
    (void)in_sizes; (void)n_in; (void)out_size; (void)ws_size;
    const float* x    = (const float*)d_in[0];
    const float* ea   = (const float*)d_in[1];
    const int*   recv = (const int*)d_in[2];
    const int*   send = (const int*)d_in[3];
    auto P = [&](int i) { return (const float*)d_in[i]; };
    // pytree (sorted-key) leaf order: dec(4..9), enc_edge(10..15), enc_node(16..21),
    // proc_edge[i](22+6i ..), proc_node[i](46+6i ..); each MLP: b0,b1,b2,w0,w1,w2

    float* ws    = (float*)d_ws;
    float* x_enc = ws;
    float* px    = x_enc + (size_t)NN * 64;
    float* agg   = px    + (size_t)NN * 64;
    float* e_enc = agg   + (size_t)NN * 64;
    float* pe    = e_enc + (size_t)NE * 64;

    const int nodeTiles = (NN + 15) / 16;
    const int edgeTiles = (NE + 15) / 16;
    const int nodeGrid  = (nodeTiles + WAVES * TPW - 1) / (WAVES * TPW);
    const int edgeGrid  = (edgeTiles + WAVES * TPW - 1) / (WAVES * TPW);
    const int aggN = NN * 64;

    encoder_kernel<NODE_F><<<nodeGrid, 128, 0, stream>>>(
        x, P(19), P(16), P(20), P(17), P(21), P(18), x_enc, px, NN);
    encoder_kernel<EDGE_F><<<edgeGrid, 128, 0, stream>>>(
        ea, P(13), P(10), P(14), P(11), P(15), P(12), e_enc, pe, NE);

    for (int s = 0; s < 4; ++s) {
        const int be = 22 + 6 * s;  // proc_edge[s]: b0,b1,b2,w0,w1,w2
        const int bn = 46 + 6 * s;  // proc_node[s]
        zero_kernel<<<(aggN + 255) / 256, 256, 0, stream>>>(agg, aggN);
        edge_block_kernel<<<edgeGrid, 128, 0, stream>>>(
            pe, px, e_enc, recv, send, agg,
            P(be + 3), P(be + 0), P(be + 4), P(be + 1), P(be + 5), P(be + 2));
        node_block_kernel<<<nodeGrid, 128, 0, stream>>>(
            px, agg, x_enc,
            P(bn + 3), P(bn + 0), P(bn + 4), P(bn + 1), P(bn + 5), P(bn + 2), NN);
    }

    decoder_kernel<<<nodeGrid, 128, 0, stream>>>(
        px, (float*)d_out, P(7), P(4), P(8), P(5), P(9), P(6), NN);
}
